// NCPDecoder_81346680586933
// MI455X (gfx1250) — compile-verified, matrix operand order checked by reading, not run on hardware
//
#include <hip/hip_runtime.h>
#include <stdint.h>

// ---------------------------------------------------------------------------
// Types for CDNA5 WMMA (wave32, 16x16x32 bf16 -> f32)
// ---------------------------------------------------------------------------
typedef __attribute__((ext_vector_type(16))) __bf16 v16bf;
typedef __attribute__((ext_vector_type(8)))  float  v8f;

union Frag { uint4 q[2]; v16bf v; };

#define B_TOT   1024
#define T_STEPS 256
#define DIN_    128
#define BM      32            // batch rows per workgroup
#define THREADS 512
#define WAVES   16

// layer geometry (K padded to x32, N padded to x16)
#define CAT0 397
#define KT0  13
#define NH0  269
#define NT0  17
#define SP0  272
#define CAT1 448
#define KT1  14
#define NH1  179
#define NT1  12
#define SP1  192
#define CAT2 243
#define KT2  8
#define NH2  64
#define NT2  4
#define SP2  64

// packed bf16 weight sizes (in ushort elements) inside d_ws
#define W0SZ (KT0*NT0*512)
#define W1SZ (KT1*NT1*512)
#define W2SZ (KT2*NT2*512)
#define OFF0 0
#define OFF1 (4*W0SZ)
#define OFF2 (OFF1 + 4*W1SZ)
#define OFFF (OFF2 + 4*W2SZ)      // fc: KT=2, NT=4 -> 4096 elems

// dynamic LDS layout (bytes)
#define S0_OFF   0
#define S0_BYTES (BM*SP0*4)
#define S1_OFF   (S0_OFF + S0_BYTES)
#define S1_BYTES (BM*SP1*4)
#define S2_OFF   (S1_OFF + S1_BYTES)
#define S2_BYTES (BM*SP2*4)
#define XA_OFF   (S2_OFF + S2_BYTES)
#define XA_BYTES (BM*128*2)
#define XC1_OFF  (XA_OFF + XA_BYTES)
#define XC1_BYTES (BM*448*2)
#define XC2_OFF  (XC1_OFF + XC1_BYTES)
#define XC2_BYTES (BM*256*2)
#define XF_OFF   (XC2_OFF + XC2_BYTES)
#define XF_BYTES (BM*64*2)
#define SMEM_BYTES (XF_OFF + XF_BYTES)   // 124,928 B < 320 KB/WGP

// ---------------------------------------------------------------------------
// helpers
// ---------------------------------------------------------------------------
__device__ __forceinline__ unsigned short f2bf(float f) {
  unsigned u = __float_as_uint(f);
  unsigned r = u + 0x7fffu + ((u >> 16) & 1u);   // round-to-nearest-even
  return (unsigned short)(r >> 16);
}

__device__ __forceinline__ float fast_tanh(float x) {
  float e = __expf(2.0f * x);
  return 1.0f - 2.0f / (e + 1.0f);
}
__device__ __forceinline__ float fast_sigmoid(float x) {
  return 1.0f / (1.0f + __expf(-x));
}

// K index inside a 32-wide tile for (lane l, element e) of a bf16 A/B fragment
// (matches the 16-bit matrix VGPR layout in CDNA5 ISA §7.12.2)
__device__ __forceinline__ int kElem(int l, int e) {
  int v = e >> 1;
  return ((v < 4) ? 0 : 16) + ((l >> 4) << 3) + ((v & 3) << 1) + (e & 1);
}

// ---------------------------------------------------------------------------
// prep kernel: f32 weights (+bool mask for ff1/ff2) -> packed bf16 B-fragments
// grid = nMat*KT*NT blocks, 512 threads; dst block = 32 lanes x 16 elems
// ---------------------------------------------------------------------------
__global__ void prep_weights(const float* __restrict__ w0,
                             const float* __restrict__ w1,
                             const float* __restrict__ w2,
                             const float* __restrict__ w3,
                             const unsigned char* __restrict__ mask,
                             int cat, int n_h, int KT, int NT,
                             unsigned short* __restrict__ dst,
                             long long matStride) {
  int blk = blockIdx.x;
  int mi  = blk / (KT * NT);
  int rem = blk % (KT * NT);
  int kt  = rem / NT, nt = rem % NT;
  int tid = threadIdx.x;
  int l = tid >> 4, e = tid & 15;
  int k = kt * 32 + kElem(l, e);
  int n = nt * 16 + (l & 15);
  const float* w = (mi == 0) ? w0 : (mi == 1) ? w1 : (mi == 2) ? w2 : w3;
  float val = 0.0f;
  if (k < cat && n < n_h) {
    val = w[(long long)n * cat + k];
    if (mask != nullptr && mi < 2 && mask[(long long)n * cat + k] == 0) val = 0.0f;
  }
  dst[mi * matStride + (long long)(kt * NT + nt) * 512 + tid] = f2bf(val);
}

// ---------------------------------------------------------------------------
// One CfC layer: 4 fused matmuls (shared A-fragment) + activation combine.
// Each wave loops over 16x16 output tiles strided by WAVES; 4 v_wmma per K-step.
// ---------------------------------------------------------------------------
template <int KT, int NT, int NH, int SP, int SSTR, bool SPLIT>
__device__ __forceinline__ void cfc_layer(
    const unsigned short* __restrict__ src,   // xc buffer, row stride SSTR elems
    const unsigned short* __restrict__ srcX,  // x buffer (stride 128) when SPLIT
    const unsigned short* __restrict__ w,     // 4 packed matrices, consecutive
    const float* __restrict__ bF1, const float* __restrict__ bF2,
    const float* __restrict__ bTA, const float* __restrict__ bTB,
    float* __restrict__ stag, int wave, int lane) {
  const int half = lane >> 4;
  const int lrow = lane & 15;
  const int MSZ  = KT * NT * 512;
  for (int tg = wave; tg < (BM / 16) * NT; tg += WAVES) {
    const int mt = tg / NT, nt = tg % NT;
    v8f c0 = {0.f,0.f,0.f,0.f,0.f,0.f,0.f,0.f};
    v8f c1 = c0, c2 = c0, c3 = c0;
    for (int kt = 0; kt < KT; ++kt) {
      // ---- A fragment from LDS: two aligned 16B loads per lane ----
      const unsigned short* abase;
      if (SPLIT && kt < 4) {
        abase = srcX + (mt * 16 + lrow) * 128 + kt * 32 + half * 8;
      } else {
        int k0 = SPLIT ? (kt * 32 - 128) : (kt * 32);
        abase = src + (mt * 16 + lrow) * SSTR + k0 + half * 8;
      }
      Frag a;
      a.q[0] = *reinterpret_cast<const uint4*>(abase);
      a.q[1] = *reinterpret_cast<const uint4*>(abase + 16);

      // ---- B fragments for ff1/ff2/ta/tb from packed weights (L2) ----
      const unsigned short* wb = w + (long long)(kt * NT + nt) * 512 + lane * 16;
      if (kt + 1 < KT)  // stream next K-tile (global_prefetch_b8)
        __builtin_prefetch(w + (long long)((kt + 1) * NT + nt) * 512 + lane * 16, 0, 3);
      Frag b0, b1, b2, b3;
      b0.q[0] = *reinterpret_cast<const uint4*>(wb);
      b0.q[1] = *reinterpret_cast<const uint4*>(wb + 8);
      b1.q[0] = *reinterpret_cast<const uint4*>(wb + MSZ);
      b1.q[1] = *reinterpret_cast<const uint4*>(wb + MSZ + 8);
      b2.q[0] = *reinterpret_cast<const uint4*>(wb + 2 * MSZ);
      b2.q[1] = *reinterpret_cast<const uint4*>(wb + 2 * MSZ + 8);
      b3.q[0] = *reinterpret_cast<const uint4*>(wb + 3 * MSZ);
      b3.q[1] = *reinterpret_cast<const uint4*>(wb + 3 * MSZ + 8);

      c0 = __builtin_amdgcn_wmma_f32_16x16x32_bf16(false, a.v, false, b0.v, (short)0, c0, false, false);
      c1 = __builtin_amdgcn_wmma_f32_16x16x32_bf16(false, a.v, false, b1.v, (short)0, c1, false, false);
      c2 = __builtin_amdgcn_wmma_f32_16x16x32_bf16(false, a.v, false, b2.v, (short)0, c2, false, false);
      c3 = __builtin_amdgcn_wmma_f32_16x16x32_bf16(false, a.v, false, b3.v, (short)0, c3, false, false);
    }
    const int n = nt * 16 + lrow;
    const bool valid = (n < NH);
    float vb0 = valid ? bF1[n] : 0.0f;
    float vb1 = valid ? bF2[n] : 0.0f;
    float vba = valid ? bTA[n] : 0.0f;
    float vbb = valid ? bTB[n] : 0.0f;
#pragma unroll
    for (int r = 0; r < 8; ++r) {
      float g1 = fast_tanh(c0[r] + vb0);
      float g2 = fast_tanh(c1[r] + vb1);
      float s  = fast_sigmoid(c2[r] + vba + c3[r] + vbb);
      float h  = g1 * (1.0f - s) + s * g2;
      int m = mt * 16 + r + half * 8;
      stag[m * SP + n] = h;
    }
  }
}

// ---------------------------------------------------------------------------
// Main persistent kernel: one workgroup owns BM batch rows for all T steps.
// ---------------------------------------------------------------------------
struct MainParams {
  const float* x;
  const float* hidden;
  const float* b00; const float* b01; const float* b02; const float* b03;
  const float* b10; const float* b11; const float* b12; const float* b13;
  const float* b20; const float* b21; const float* b22; const float* b23;
  const float* fcb;
  const unsigned short* wts;
  float* out;
};

__global__ void __launch_bounds__(THREADS, 1) ncp_main(MainParams p) {
  extern __shared__ char smem[];
  float* s0  = reinterpret_cast<float*>(smem + S0_OFF);
  float* s1  = reinterpret_cast<float*>(smem + S1_OFF);
  float* s2  = reinterpret_cast<float*>(smem + S2_OFF);
  unsigned short* xA  = reinterpret_cast<unsigned short*>(smem + XA_OFF);
  unsigned short* xc1 = reinterpret_cast<unsigned short*>(smem + XC1_OFF);
  unsigned short* xc2 = reinterpret_cast<unsigned short*>(smem + XC2_OFF);
  unsigned short* xf  = reinterpret_cast<unsigned short*>(smem + XF_OFF);

  const int tid  = threadIdx.x;
  const int wave = tid >> 5;
  const int lane = tid & 31;
  const int rowBase = blockIdx.x * BM;

  // seed hidden state: h0 -> xc1[:,0:269], h1 -> xc1[:,269:448], h2 -> xc2[:,179:243]
  for (int idx = tid; idx < BM * 512; idx += THREADS) {
    int m = idx >> 9, j = idx & 511;
    unsigned short hb = f2bf(p.hidden[(long long)(rowBase + m) * 512 + j]);
    if (j < 448) xc1[m * 448 + j] = hb;
    else         xc2[m * 256 + 179 + (j - 448)] = hb;
  }
  __syncthreads();

  for (int t = 0; t < T_STEPS; ++t) {
    // stage x_t (f32 -> bf16)
    for (int idx = tid; idx < BM * DIN_; idx += THREADS) {
      int m = idx >> 7, k = idx & 127;
      xA[m * 128 + k] =
          f2bf(p.x[((long long)(rowBase + m) * T_STEPS + t) * DIN_ + k]);
    }
    __syncthreads();

    // ---- layer 0: xc = [x_t | h0] ----
    cfc_layer<KT0, NT0, NH0, SP0, 448, true>(xc1, xA, p.wts + OFF0,
                                             p.b00, p.b01, p.b02, p.b03,
                                             s0, wave, lane);
    __syncthreads();
    for (int idx = tid; idx < BM * NH0; idx += THREADS) {
      int m = idx / NH0, j = idx % NH0;
      xc1[m * 448 + j] = f2bf(s0[m * SP0 + j]);          // n0 -> xc1[:,0:269]
    }
    __syncthreads();

    // ---- layer 1: xc = [n0 | h1] ----
    cfc_layer<KT1, NT1, NH1, SP1, 448, false>(xc1, nullptr, p.wts + OFF1,
                                              p.b10, p.b11, p.b12, p.b13,
                                              s1, wave, lane);
    __syncthreads();
    for (int idx = tid; idx < BM * NH1; idx += THREADS) {
      int m = idx / NH1, j = idx % NH1;
      unsigned short v = f2bf(s1[m * SP1 + j]);
      xc2[m * 256 + j]       = v;                        // n1 -> xc2[:,0:179]
      xc1[m * 448 + NH0 + j] = v;                        // n1 -> h1 slot for t+1
    }
    __syncthreads();

    // ---- layer 2: xc = [n1 | h2] ----
    cfc_layer<KT2, NT2, NH2, SP2, 256, false>(xc2, nullptr, p.wts + OFF2,
                                              p.b20, p.b21, p.b22, p.b23,
                                              s2, wave, lane);
    __syncthreads();
    for (int idx = tid; idx < BM * 64; idx += THREADS) {
      int m = idx >> 6, j = idx & 63;
      unsigned short v = f2bf(s2[m * 64 + j]);
      xc2[m * 256 + NH1 + j] = v;                        // n2 -> h2 slot for t+1
      xf[m * 64 + j]         = v;                        // n2 -> FC input
    }
    __syncthreads();

    // ---- FC head: predictions[b,t,:] = n2 @ fc_w.T + fc_b (WMMA, K=64) ----
    if (wave < (BM / 16) * 4) {
      const int mt = wave >> 2, nt = wave & 3;
      const int half = lane >> 4, lrow = lane & 15;
      v8f c = {0.f,0.f,0.f,0.f,0.f,0.f,0.f,0.f};
#pragma unroll
      for (int kt = 0; kt < 2; ++kt) {
        const unsigned short* abase = xf + (mt * 16 + lrow) * 64 + kt * 32 + half * 8;
        Frag a, b;
        a.q[0] = *reinterpret_cast<const uint4*>(abase);
        a.q[1] = *reinterpret_cast<const uint4*>(abase + 16);
        const unsigned short* wb = p.wts + OFFF + (kt * 4 + nt) * 512 + lane * 16;
        b.q[0] = *reinterpret_cast<const uint4*>(wb);
        b.q[1] = *reinterpret_cast<const uint4*>(wb + 8);
        c = __builtin_amdgcn_wmma_f32_16x16x32_bf16(false, a.v, false, b.v, (short)0, c, false, false);
      }
      const int n = nt * 16 + lrow;
      const float fb = p.fcb[n];
#pragma unroll
      for (int r = 0; r < 8; ++r) {
        int m = mt * 16 + r + half * 8;
        long long gb = rowBase + m;
        p.out[(gb * T_STEPS + t) * 64 + n] = c[r] + fb;
      }
    }
    __syncthreads();
  }

  // final hidden state hn = [f0 | f1 | f2] (f32, from staging of last step)
  for (int idx = tid; idx < BM * 512; idx += THREADS) {
    int m = idx >> 9, j = idx & 511;
    float v = (j < NH0) ? s0[m * SP0 + j]
            : (j < NH0 + NH1) ? s1[m * SP1 + (j - NH0)]
            : s2[m * 64 + (j - NH0 - NH1)];
    p.out[(long long)B_TOT * T_STEPS * 64 + (long long)(rowBase + m) * 512 + j] = v;
  }
}

// ---------------------------------------------------------------------------
// launcher
// ---------------------------------------------------------------------------
extern "C" void kernel_launch(void* const* d_in, const int* in_sizes, int n_in,
                              void* d_out, int out_size, void* d_ws, size_t ws_size,
                              hipStream_t stream) {
  (void)in_sizes; (void)n_in; (void)out_size; (void)ws_size;
  const float* x      = (const float*)d_in[0];
  const float* hidden = (const float*)d_in[1];
  const unsigned char* mask0 = (const unsigned char*)d_in[2];
  const unsigned char* mask1 = (const unsigned char*)d_in[3];
  const unsigned char* mask2 = (const unsigned char*)d_in[4];
  // per layer: ff1_w, ff1_b, ff2_w, ff2_b, ta_w, ta_b, tb_w, tb_b
  const float* ff1_w0 = (const float*)d_in[5];  const float* ff1_b0 = (const float*)d_in[6];
  const float* ff2_w0 = (const float*)d_in[7];  const float* ff2_b0 = (const float*)d_in[8];
  const float* ta_w0  = (const float*)d_in[9];  const float* ta_b0  = (const float*)d_in[10];
  const float* tb_w0  = (const float*)d_in[11]; const float* tb_b0  = (const float*)d_in[12];
  const float* ff1_w1 = (const float*)d_in[13]; const float* ff1_b1 = (const float*)d_in[14];
  const float* ff2_w1 = (const float*)d_in[15]; const float* ff2_b1 = (const float*)d_in[16];
  const float* ta_w1  = (const float*)d_in[17]; const float* ta_b1  = (const float*)d_in[18];
  const float* tb_w1  = (const float*)d_in[19]; const float* tb_b1  = (const float*)d_in[20];
  const float* ff1_w2 = (const float*)d_in[21]; const float* ff1_b2 = (const float*)d_in[22];
  const float* ff2_w2 = (const float*)d_in[23]; const float* ff2_b2 = (const float*)d_in[24];
  const float* ta_w2  = (const float*)d_in[25]; const float* ta_b2  = (const float*)d_in[26];
  const float* tb_w2  = (const float*)d_in[27]; const float* tb_b2  = (const float*)d_in[28];
  const float* fc_w   = (const float*)d_in[29];
  const float* fc_b   = (const float*)d_in[30];

  unsigned short* wts = (unsigned short*)d_ws;

  prep_weights<<<4 * KT0 * NT0, 512, 0, stream>>>(ff1_w0, ff2_w0, ta_w0, tb_w0,
      mask0, CAT0, NH0, KT0, NT0, wts + OFF0, (long long)W0SZ);
  prep_weights<<<4 * KT1 * NT1, 512, 0, stream>>>(ff1_w1, ff2_w1, ta_w1, tb_w1,
      mask1, CAT1, NH1, KT1, NT1, wts + OFF1, (long long)W1SZ);
  prep_weights<<<4 * KT2 * NT2, 512, 0, stream>>>(ff1_w2, ff2_w2, ta_w2, tb_w2,
      mask2, CAT2, NH2, KT2, NT2, wts + OFF2, (long long)W2SZ);
  prep_weights<<<2 * 4, 512, 0, stream>>>(fc_w, fc_w, fc_w, fc_w,
      nullptr, 64, 64, 2, 4, wts + OFFF, 0LL);

  MainParams mp;
  mp.x = x; mp.hidden = hidden;
  mp.b00 = ff1_b0; mp.b01 = ff2_b0; mp.b02 = ta_b0; mp.b03 = tb_b0;
  mp.b10 = ff1_b1; mp.b11 = ff2_b1; mp.b12 = ta_b1; mp.b13 = tb_b1;
  mp.b20 = ff1_b2; mp.b21 = ff2_b2; mp.b22 = ta_b2; mp.b23 = tb_b2;
  mp.fcb = fc_b; mp.wts = wts; mp.out = (float*)d_out;

  ncp_main<<<B_TOT / BM, THREADS, SMEM_BYTES, stream>>>(mp);
}
